// GNNActor_89026082111547
// MI455X (gfx1250) — matrix operand order repaired and checked by reading, not compiled, not run on hardware
//
#include <hip/hip_runtime.h>
#include <hip/hip_bf16.h>
#include <stdint.h>

typedef __bf16 bf16_t;
typedef __attribute__((ext_vector_type(16))) __bf16 v16bf;
typedef __attribute__((ext_vector_type(8)))  float  v8f;

#define IN_DIM  64
#define HID     128
#define OUT_DIM 16

// ---------------- utility kernels ----------------
__global__ void k_zero(float* __restrict__ p, long n) {
    long i = (long)blockIdx.x * blockDim.x + threadIdx.x;
    if (i < n) p[i] = 0.f;
}

__global__ void k_init_deg(float* __restrict__ deg, int n) {
    int i = blockIdx.x * blockDim.x + threadIdx.x;
    if (i < n) deg[i] = 1.0f;   // self-loop contributes 1
}

__global__ void k_deg_accum(const long long* __restrict__ ei, float* __restrict__ deg, int E) {
    int e = blockIdx.x * blockDim.x + threadIdx.x;
    if (e < E) atomicAdd(&deg[(int)ei[(long)E + e]], 1.0f);  // dst = row 1
}

__global__ void k_rsqrt(float* __restrict__ d, int n) {
    int i = blockIdx.x * blockDim.x + threadIdx.x;
    if (i < n) d[i] = rsqrtf(d[i]);
}

__global__ void k_cvt_bf16(const float* __restrict__ x, bf16_t* __restrict__ y, long n) {
    long i = (long)blockIdx.x * blockDim.x + threadIdx.x;
    if (i < n) y[i] = (bf16_t)x[i];
}

// W [K x Ncols] row-major fp32  ->  Wt [Ncols x K] row-major bf16 (transposed)
__global__ void k_cvt_w_t(const float* __restrict__ W, bf16_t* __restrict__ Wt, int K, int Ncols) {
    int i = blockIdx.x * blockDim.x + threadIdx.x;
    if (i < K * Ncols) {
        int k = i / Ncols, n = i % Ncols;
        Wt[n * K + k] = (bf16_t)W[i];
    }
}

// ---------------- WMMA GEMM: H[M x 128] = A[M x K](bf16) @ W[K x 128](bf16, stored transposed) ----------------
// block = 256 threads = 8 waves; block computes 16 rows x 128 cols; wave w -> cols [16w,16w+16)
__global__ __launch_bounds__(256) void k_gemm_bf16(
    const bf16_t* __restrict__ A, const bf16_t* __restrict__ Wt,
    float* __restrict__ H, int K)
{
    __shared__ __align__(16) bf16_t ldsA[16 * HID];  // 16 rows x K bf16 (max 4 KB)

    const int tid   = threadIdx.x;
    const int tileM = blockIdx.x;

    // Stage A tile: 16 consecutive rows are contiguous -> straight coalesced uint4 copy
    {
        const uint4* src = (const uint4*)(A + (size_t)tileM * 16 * K);
        uint4*       dst = (uint4*)ldsA;
        const int nchunks = (16 * K) >> 3;           // 128 (K=64) or 256 (K=128)
        for (int c = tid; c < nchunks; c += 256) dst[c] = src[c];
    }
    __syncthreads();

    const int wave = tid >> 5;
    const int lane = tid & 31;
    const int half = lane >> 4;      // lane group 0: lanes 0-15, 1: lanes 16-31
    const int ml   = lane & 15;
    const int n    = wave * 16 + ml; // output column

    v8f acc = {0.f, 0.f, 0.f, 0.f, 0.f, 0.f, 0.f, 0.f};

    const uint4* lq = (const uint4*)ldsA;
    const uint4* wq = (const uint4*)Wt;
    const int kb_count = K >> 5;
    const int aRow = (ml * K) >> 3;  // uint4 index of this lane's A row
    const int bRow = (n  * K) >> 3;  // uint4 index of this lane's W column (transposed storage)

    for (int kb = 0; kb < kb_count; ++kb) {
        // A 16x32 bf16 fragment (ISA layout): e=0..7 -> K = 32kb + 8*half + e ; e=8..15 -> +16
        union { uint4 q[2]; v16bf v; } af;
        af.q[0] = lq[aRow + kb * 4 + half];
        af.q[1] = lq[aRow + kb * 4 + half + 2];
        // B 32x16 bf16 fragment: e=0..7 -> K = 32kb + 8*half + e ; e=8..15 -> +16 ; col N = ml
        union { uint4 q[2]; v16bf v; } bfm;
        bfm.q[0] = wq[bRow + kb * 4 + half];
        bfm.q[1] = wq[bRow + kb * 4 + half + 2];

        acc = __builtin_amdgcn_wmma_f32_16x16x32_bf16(
            /*neg_a=*/false, af.v, /*neg_b=*/false, bfm.v,
            /*c_mod=*/(short)0, acc, /*reuse_a=*/false, /*reuse_b=*/false);
    }

    // C/D layout: VGPR r -> M = r + 8*half, N = ml
    float* out = H + ((size_t)tileM * 16 + half * 8) * HID + n;
#pragma unroll
    for (int r = 0; r < 8; ++r) out[(size_t)r * HID] = acc[r];
}

// ---------------- normalized scatter-add over edges (incl. self-loops e>=E) ----------------
__global__ __launch_bounds__(256) void k_aggregate(
    const float* __restrict__ h, const long long* __restrict__ ei,
    const float* __restrict__ dinv, float* __restrict__ agg,
    int E, int N, int ET, int total)
{
    int gw = blockIdx.x * 8 + (threadIdx.x >> 5);   // one wave per (batch, edge)
    if (gw >= total) return;
    int lane = threadIdx.x & 31;
    int b = gw / ET;
    int e = gw - b * ET;

    int s, d; float nrm;
    if (e < E) {
        s = (int)ei[e];            // src row 0
        d = (int)ei[(long)E + e];  // dst row 1
        nrm = dinv[s] * dinv[d];
    } else {
        s = d = e - E;             // self-loop
        float di = dinv[s];
        nrm = di * di;
    }

    const float4 hv = ((const float4*)(h + ((size_t)b * N + s) * HID))[lane];
    float* dst = agg + ((size_t)b * N + d) * HID + lane * 4;
    atomicAdd(dst + 0, nrm * hv.x);
    atomicAdd(dst + 1, nrm * hv.y);
    atomicAdd(dst + 2, nrm * hv.z);
    atomicAdd(dst + 3, nrm * hv.w);
}

// ---------------- bias + relu + convert to bf16 (layer-2 input) ----------------
__global__ void k_bias_relu_cvt(const float* __restrict__ agg, const float* __restrict__ bias,
                                bf16_t* __restrict__ y, long total)
{
    long i = (long)blockIdx.x * blockDim.x + threadIdx.x;
    if (i < total) {
        float v = agg[i] + bias[i & (HID - 1)];
        y[i] = (bf16_t)fmaxf(v, 0.f);
    }
}

// ---------------- bias + relu + mean-pool partial reduction ----------------
// grid (chunks, B), block 128; thread = hidden dim
__global__ void k_pool(const float* __restrict__ agg, const float* __restrict__ b2,
                       float* __restrict__ pool, int N, int per)
{
    int b  = blockIdx.y;
    int hh = threadIdx.x;
    int n0 = blockIdx.x * per;
    int n1 = min(N, n0 + per);
    float bias = b2[hh], s = 0.f;
    for (int n = n0; n < n1; ++n)
        s += fmaxf(agg[((size_t)b * N + n) * HID + hh] + bias, 0.f);
    atomicAdd(&pool[b * HID + hh], s);
}

// ---------------- FC head + sigma ----------------
__global__ void k_final(const float* __restrict__ pool, const float* __restrict__ addf,
                        const float* __restrict__ fcw, const float* __restrict__ fcb,
                        const float* __restrict__ sp, float* __restrict__ out, int N, int B)
{
    int t = threadIdx.x;
    if (t >= B * OUT_DIM) return;
    int b = t / OUT_DIM, o = t % OUT_DIM;
    float invN = 1.f / (float)N;
    float acc = fcb[o] + addf[b] * fcw[HID * OUT_DIM + o];  // last fc_w row x additional feature
    for (int k = 0; k < HID; ++k)
        acc += pool[b * HID + k] * invN * fcw[k * OUT_DIM + o];
    out[t] = acc;                          // mu [B,OUT]
    out[B * OUT_DIM + t] = expf(sp[o]);    // sigma [B,OUT] = exp(sigma_param) broadcast
}

// ---------------- launch ----------------
extern "C" void kernel_launch(void* const* d_in, const int* in_sizes, int n_in,
                              void* d_out, int out_size, void* d_ws, size_t ws_size,
                              hipStream_t stream)
{
    const float*     x    = (const float*)d_in[0];      // [B,N,64]
    const float*     addf = (const float*)d_in[1];      // [B,1]
    const long long* ei   = (const long long*)d_in[2];  // [2,E] int64
    const float*     W1   = (const float*)d_in[3];      // [64,128]
    const float*     b1   = (const float*)d_in[4];
    const float*     W2   = (const float*)d_in[5];      // [128,128]
    const float*     b2   = (const float*)d_in[6];
    const float*     fcw  = (const float*)d_in[7];      // [129,16]
    const float*     fcb  = (const float*)d_in[8];
    const float*     sp   = (const float*)d_in[9];      // [16,1]
    float* out = (float*)d_out;

    const int B = 4, N = 20000;
    const int E = in_sizes[2] / 2;
    const long MN   = (long)B * N;          // 80000 rows
    const long aggN = MN * HID;             // 10,240,000

    // workspace layout (256B aligned slices)
    char* ws = (char*)d_ws; size_t off = 0;
    auto wsalloc = [&](size_t bytes) { char* p = ws + off; off = (off + bytes + 255) & ~(size_t)255; return p; };
    float*  dinv = (float*) wsalloc((size_t)N * 4);
    bf16_t* xb   = (bf16_t*)wsalloc((size_t)MN * IN_DIM * 2);
    bf16_t* W1t  = (bf16_t*)wsalloc((size_t)IN_DIM * HID * 2);
    bf16_t* W2t  = (bf16_t*)wsalloc((size_t)HID * HID * 2);
    float*  h    = (float*) wsalloc((size_t)aggN * 4);   // GEMM output, reused both layers
    float*  agg  = (float*) wsalloc((size_t)aggN * 4);   // aggregation target, re-zeroed
    bf16_t* x2b  = (bf16_t*)wsalloc((size_t)aggN * 2);
    float*  pool = (float*) wsalloc((size_t)B * HID * 4);
    (void)ws_size; (void)n_in; (void)out_size;

    const int ET = E + N;            // edges + self-loops
    const int tasks = B * ET;        // one wave each

    // degree / norm
    k_init_deg <<<(N + 255) / 256, 256, 0, stream>>>(dinv, N);
    k_deg_accum<<<(E + 255) / 256, 256, 0, stream>>>(ei, dinv, E);
    k_rsqrt    <<<(N + 255) / 256, 256, 0, stream>>>(dinv, N);

    // zero accumulators
    k_zero<<<(int)((aggN + 255) / 256), 256, 0, stream>>>(agg, aggN);
    k_zero<<<(B * HID + 255) / 256, 256, 0, stream>>>(pool, B * HID);

    // bf16 conversions (weights transposed for contiguous B-fragment loads)
    k_cvt_bf16<<<(int)((MN * IN_DIM + 255) / 256), 256, 0, stream>>>(x, xb, MN * IN_DIM);
    k_cvt_w_t <<<(IN_DIM * HID + 255) / 256, 256, 0, stream>>>(W1, W1t, IN_DIM, HID);
    k_cvt_w_t <<<(HID * HID + 255) / 256, 256, 0, stream>>>(W2, W2t, HID, HID);

    // layer 1: GEMM -> scatter -> bias/relu/cvt
    k_gemm_bf16<<<(int)(MN / 16), 256, 0, stream>>>(xb, W1t, h, IN_DIM);
    k_aggregate<<<(tasks + 7) / 8, 256, 0, stream>>>(h, ei, dinv, agg, E, N, ET, tasks);
    k_bias_relu_cvt<<<(int)((aggN + 255) / 256), 256, 0, stream>>>(agg, b1, x2b, aggN);

    // layer 2: re-zero -> GEMM -> scatter -> pool
    k_zero<<<(int)((aggN + 255) / 256), 256, 0, stream>>>(agg, aggN);
    k_gemm_bf16<<<(int)(MN / 16), 256, 0, stream>>>(x2b, W2t, h, HID);
    k_aggregate<<<(tasks + 7) / 8, 256, 0, stream>>>(h, ei, dinv, agg, E, N, ET, tasks);

    const int chunks = 64;
    const int per = (N + chunks - 1) / chunks;
    k_pool <<<dim3(chunks, B), 128, 0, stream>>>(agg, b2, pool, N, per);
    k_final<<<1, 64, 0, stream>>>(pool, addf, fcw, fcb, sp, out, N, B);
}
